// NonLocal_15006615733543
// MI455X (gfx1250) — matrix operand order repaired and checked by reading, not compiled, not run on hardware
//
#include <hip/hip_runtime.h>

#define DEVINL __device__ __forceinline__

typedef __attribute__((ext_vector_type(16))) __bf16 v16bf;
typedef __attribute__((ext_vector_type(8)))  __bf16 v8bf;
typedef __attribute__((ext_vector_type(8)))  float  v8f;
typedef __attribute__((ext_vector_type(4)))  unsigned v4u;
typedef __attribute__((ext_vector_type(8)))  unsigned v8u;

constexpr int CB   = 2;       // batch
constexpr int CIN  = 256;     // input channels
constexpr int CI   = 128;     // intermediate channels (d)
constexpr int NTOK = 8192;    // 8*32*32 tokens

DEVINL __bf16 f2bf(float f) { return (__bf16)f; }   // native cvt on gfx1250

DEVINL v16bf cat8(v8bf lo, v8bf hi) {
  return __builtin_shufflevector(lo, hi, 0,1,2,3,4,5,6,7,8,9,10,11,12,13,14,15);
}

// ---------------------------------------------------------------------------
// Kernel 1: projections  theta/phi/g = X^T @ W^T + b   -> bf16 [b, token, d]
// One wave per 16(token) x 16(d) tile; K = 256 in 8 WMMA steps.
// grid (NTOK/16, CI/16, 3*CB), block 32
// ---------------------------------------------------------------------------
__global__ __launch_bounds__(32) void proj_kernel(
    const float* __restrict__ x,
    const float* __restrict__ gw, const float* __restrict__ gb,
    const float* __restrict__ tw, const float* __restrict__ tb,
    const float* __restrict__ pw, const float* __restrict__ pb,
    __bf16* __restrict__ g_o, __bf16* __restrict__ t_o, __bf16* __restrict__ p_o)
{
  const int lane = threadIdx.x;
  const bool hiL = lane >= 16;
  const int l15  = lane & 15;
  const int n0   = blockIdx.x * 16;
  const int d0   = blockIdx.y * 16;
  const int b    = blockIdx.z & 1;
  const int proj = blockIdx.z >> 1;

  const float* W; const float* bias; __bf16* out;
  if (proj == 0)      { W = gw; bias = gb; out = g_o; }
  else if (proj == 1) { W = tw; bias = tb; out = t_o; }
  else                { W = pw; bias = pb; out = p_o; }

  const float bv = bias[d0 + l15];
  v8f acc;
#pragma unroll
  for (int r = 0; r < 8; ++r) acc[r] = bv;

  const float* xb   = x + (size_t)b * CIN * NTOK + n0 + l15;   // column n0+l15
  const float* wrow = W + (size_t)(d0 + l15) * CIN;            // row d0+l15

#pragma unroll
  for (int kc = 0; kc < 8; ++kc) {          // K chunks of 32 over CIN
    const int k0 = kc * 32;
    v16bf a;                                // A: M=token, K=channel
#pragma unroll
    for (int i = 0; i < 8; ++i) {
      const int cA = k0 + (hiL ? 16 : 0) + i;       // K 0..7 / 16..23
      const int cB = k0 + 8 + (hiL ? 16 : 0) + i;   // K 8..15 / 24..31
      a[i]     = f2bf(xb[(size_t)cA * NTOK]);
      a[i + 8] = f2bf(xb[(size_t)cB * NTOK]);
    }
    v16bf bfr;                              // B: K=channel, N=d (lane)
    const float* wp = wrow + k0 + (hiL ? 16 : 0);
#pragma unroll
    for (int i = 0; i < 16; ++i) bfr[i] = f2bf(wp[i]);
    acc = __builtin_amdgcn_wmma_f32_16x16x32_bf16(false, a, false, bfr,
                                                  (short)0, acc, false, false);
  }

#pragma unroll
  for (int r = 0; r < 8; ++r) {
    const int tok = n0 + r + (hiL ? 8 : 0);
    out[((size_t)b * NTOK + tok) * CI + d0 + l15] = f2bf(acc[r]);
  }
}

// ---------------------------------------------------------------------------
// Kernel 2: flash attention.  8 waves/block, 16 queries/wave, key tiles of 32.
// K tile staged by the Tensor Data Mover (tensor_load_to_lds) with TDM padding
// reproducing the 272B conflict-free LDS row stride.  V staged transposed.
// grid (NTOK/128, CB), block 256
// ---------------------------------------------------------------------------
__global__ __launch_bounds__(256) void attn_kernel(
    const __bf16* __restrict__ Tq,   // theta  [b, token, d]
    const __bf16* __restrict__ Pk,   // phi    [b, token, d] (keys)
    const __bf16* __restrict__ Gv,   // g      [b, token, d] (values)
    __bf16* __restrict__ Y)          // out    [b, token, d]
{
  constexpr int KPAD = 272;  // bytes per K-tile row (128 bf16 + 16B pad)
  constexpr int VPAD = 80;   // bytes per Vt row / P row (32 bf16 + pad)
  __shared__ __align__(16) unsigned char ldsK[32 * KPAD];       // K tile [key][d]
  __shared__ __align__(16) unsigned char ldsV[128 * VPAD];      // V tile transposed [d][key]
  __shared__ __align__(16) unsigned char ldsP[8 * 16 * VPAD];   // per-wave P staging [q][k]

  const int tid  = threadIdx.x;
  const int lane = tid & 31;
  const int wave = tid >> 5;
  const bool hiL = lane >= 16;
  const int l15  = lane & 15;
  const int b    = blockIdx.y;
  const int q0   = blockIdx.x * 128 + wave * 16;

  // Q fragments (A layout), constant over key loop
  const __bf16* qrow = Tq + ((size_t)b * NTOK + q0 + l15) * CI;
  v16bf Qf[4];
#pragma unroll
  for (int c = 0; c < 4; ++c) {
    const __bf16* p = qrow + c * 32 + (hiL ? 16 : 0);
    Qf[c] = cat8(*(const v8bf*)p, *(const v8bf*)(p + 8));
  }

  v8f O[8];
  float mr[8], lr[8];
#pragma unroll
  for (int t = 0; t < 8; ++t)
#pragma unroll
    for (int r = 0; r < 8; ++r) O[t][r] = 0.f;
#pragma unroll
  for (int r = 0; r < 8; ++r) { mr[r] = -1e30f; lr[r] = 0.f; }

  const int key = tid >> 3;          // cooperative V-loader role: 0..31
  const int dbl = (tid & 7) * 16;    // 0..112
  const unsigned ldsKoff = (unsigned)(unsigned long long)(void*)ldsK;

  for (int k0 = 0; k0 < NTOK; k0 += 32) {
    __syncthreads();

    // ---- TDM: DMA the 32x128 bf16 K tile into LDS with 16B row padding ----
    if (wave == 0) {
      const unsigned long long ga =
          (unsigned long long)(const void*)(Pk + ((size_t)b * NTOK + k0) * CI);
      v4u g0;
      g0[0] = 1u;                                 // count=1, user descriptor
      g0[1] = ldsKoff;                            // lds_addr
      g0[2] = (unsigned)ga;                       // global_addr[31:0]
      g0[3] = (unsigned)(ga >> 32) | (2u << 30);  // global_addr[56:32] | type=2
      v8u g1;
      g1[0] = (1u << 16)        // data_size = 2 bytes
            | (1u << 20)        // pad_enable
            | (5u << 22)        // pad_interval: every 64 DWORDs (256B row)
            | (3u << 25);       // pad_amount: 4 DWORDs (16B) -> 272B stride
      g1[1] = (unsigned)CI << 16;   // tensor_dim0 = 128 elements
      g1[2] = 32u << 16;            // tensor_dim1 = 32 rows
      g1[3] = (unsigned)CI << 16;   // tile_dim0 = 128
      g1[4] = 32u;                  // tile_dim1 = 32, tile_dim2 = 0
      g1[5] = (unsigned)CI;         // tensor_dim0_stride = 128
      g1[6] = 0u;
      g1[7] = 0u;
      asm volatile("tensor_load_to_lds %0, %1" :: "s"(g0), "s"(g1) : "memory");
#if __has_builtin(__builtin_amdgcn_s_wait_tensorcnt)
      __builtin_amdgcn_s_wait_tensorcnt(0);
#else
      asm volatile("s_wait_tensorcnt 0x0" ::: "memory");
#endif
    }

    // ---- stage V transposed: ldsV[d][key] ----
    {
      const __bf16* vs = Gv + ((size_t)b * NTOK + k0 + key) * CI + dbl;
      v8bf v0 = *(const v8bf*)vs;
      v8bf v1 = *(const v8bf*)(vs + 8);
#pragma unroll
      for (int i = 0; i < 8; ++i) {
        *(__bf16*)(ldsV + (dbl + i) * VPAD + key * 2)     = v0[i];
        *(__bf16*)(ldsV + (dbl + 8 + i) * VPAD + key * 2) = v1[i];
      }
      if (k0 + 32 < NTOK) __builtin_prefetch(vs + 32 * CI, 0, 0);
    }
    __syncthreads();

    // S = Q @ K^T   (16 x 32, two 16x16 C tiles)
    v8f S0, S1;
#pragma unroll
    for (int r = 0; r < 8; ++r) { S0[r] = 0.f; S1[r] = 0.f; }
#pragma unroll
    for (int c = 0; c < 4; ++c) {
      const __bf16* kp0 = (const __bf16*)(ldsK + l15 * KPAD)        + c * 32 + (hiL ? 16 : 0);
      const __bf16* kp1 = (const __bf16*)(ldsK + (16 + l15) * KPAD) + c * 32 + (hiL ? 16 : 0);
      v16bf B0 = cat8(*(const v8bf*)kp0, *(const v8bf*)(kp0 + 8));
      v16bf B1 = cat8(*(const v8bf*)kp1, *(const v8bf*)(kp1 + 8));
      S0 = __builtin_amdgcn_wmma_f32_16x16x32_bf16(false, Qf[c], false, B0, (short)0, S0, false, false);
      S1 = __builtin_amdgcn_wmma_f32_16x16x32_bf16(false, Qf[c], false, B1, (short)0, S1, false, false);
    }

    // online softmax per row (rows live in 16-lane halves; xor<16 stays in-half)
    unsigned char* pbase = ldsP + wave * 16 * VPAD;
#pragma unroll
    for (int r = 0; r < 8; ++r) {
      float mx = fmaxf(S0[r], S1[r]);
#pragma unroll
      for (int off = 1; off < 16; off <<= 1)
        mx = fmaxf(mx, __shfl_xor(mx, off, 32));
      const float mnew = fmaxf(mr[r], mx);
      const float corr = __expf(mr[r] - mnew);
      mr[r] = mnew;
      const float p0 = __expf(S0[r] - mnew);
      const float p1 = __expf(S1[r] - mnew);
      float rs = p0 + p1;
#pragma unroll
      for (int off = 1; off < 16; off <<= 1)
        rs += __shfl_xor(rs, off, 32);
      lr[r] = lr[r] * corr + rs;
#pragma unroll
      for (int t = 0; t < 8; ++t) O[t][r] *= corr;
      __bf16* prow = (__bf16*)(pbase + (r + (hiL ? 8 : 0)) * VPAD);
      prow[l15]      = f2bf(p0);
      prow[16 + l15] = f2bf(p1);
    }

    // read P back in A-fragment layout (wave-private LDS; dscnt-ordered)
    const __bf16* pp = (const __bf16*)(pbase + l15 * VPAD) + (hiL ? 16 : 0);
    v16bf Pf = cat8(*(const v8bf*)pp, *(const v8bf*)(pp + 8));

    // O += P @ V
#pragma unroll
    for (int t = 0; t < 8; ++t) {
      const __bf16* vp = (const __bf16*)(ldsV + (t * 16 + l15) * VPAD) + (hiL ? 16 : 0);
      v16bf Vf = cat8(*(const v8bf*)vp, *(const v8bf*)(vp + 8));
      O[t] = __builtin_amdgcn_wmma_f32_16x16x32_bf16(false, Pf, false, Vf, (short)0, O[t], false, false);
    }
  }

  // epilogue: Y = O / l
#pragma unroll
  for (int r = 0; r < 8; ++r) {
    const float inv = 1.f / lr[r];
    const int tok = q0 + r + (hiL ? 8 : 0);
    __bf16* yo = Y + ((size_t)b * NTOK + tok) * CI + l15;
#pragma unroll
    for (int t = 0; t < 8; ++t) yo[t * 16] = f2bf(O[t][r] * inv);
  }
}

// ---------------------------------------------------------------------------
// Kernel 3: out = Y @ w_w^T + w_b + x   (fp32 output, [b, c, token] layout)
// grid (NTOK/16, CIN/16, CB), block 32
// ---------------------------------------------------------------------------
__global__ __launch_bounds__(32) void out_kernel(
    const __bf16* __restrict__ Y, const float* __restrict__ ww,
    const float* __restrict__ wb, const float* __restrict__ x,
    float* __restrict__ out)
{
  const int lane = threadIdx.x;
  const bool hiL = lane >= 16;
  const int l15  = lane & 15;
  const int n0   = blockIdx.x * 16;
  const int c0   = blockIdx.y * 16;
  const int b    = blockIdx.z;

  const float bv = wb[c0 + l15];
  v8f acc;
#pragma unroll
  for (int r = 0; r < 8; ++r) acc[r] = bv;

  const __bf16* yrow = Y + ((size_t)b * NTOK + n0 + l15) * CI;
  const float* wrow  = ww + (size_t)(c0 + l15) * CI;
#pragma unroll
  for (int c = 0; c < 4; ++c) {
    const __bf16* p = yrow + c * 32 + (hiL ? 16 : 0);
    v16bf A = cat8(*(const v8bf*)p, *(const v8bf*)(p + 8));
    v16bf Bv;
    const float* wp = wrow + c * 32 + (hiL ? 16 : 0);
#pragma unroll
    for (int i = 0; i < 16; ++i) Bv[i] = f2bf(wp[i]);
    acc = __builtin_amdgcn_wmma_f32_16x16x32_bf16(false, A, false, Bv,
                                                  (short)0, acc, false, false);
  }

#pragma unroll
  for (int r = 0; r < 8; ++r) {
    const int tok = n0 + r + (hiL ? 8 : 0);
    const size_t idx = ((size_t)b * CIN + c0 + l15) * NTOK + tok;
    out[idx] = acc[r] + x[idx];
  }
}

// ---------------------------------------------------------------------------
extern "C" void kernel_launch(void* const* d_in, const int* in_sizes, int n_in,
                              void* d_out, int out_size, void* d_ws, size_t ws_size,
                              hipStream_t stream) {
  const float* x       = (const float*)d_in[0];
  const float* g_w     = (const float*)d_in[1];
  const float* g_b     = (const float*)d_in[2];
  const float* theta_w = (const float*)d_in[3];
  const float* theta_b = (const float*)d_in[4];
  const float* phi_w   = (const float*)d_in[5];
  const float* phi_b   = (const float*)d_in[6];
  const float* w_w     = (const float*)d_in[7];
  const float* w_b     = (const float*)d_in[8];

  char* ws = (char*)d_ws;
  const size_t ACT = (size_t)CB * NTOK * CI * sizeof(__bf16);  // 4 MiB each
  __bf16* gbf = (__bf16*)(ws);
  __bf16* tbf = (__bf16*)(ws + ACT);
  __bf16* pbf = (__bf16*)(ws + 2 * ACT);
  __bf16* ybf = (__bf16*)(ws + 3 * ACT);

  proj_kernel<<<dim3(NTOK / 16, CI / 16, 3 * CB), 32, 0, stream>>>(
      x, g_w, g_b, theta_w, theta_b, phi_w, phi_b, gbf, tbf, pbf);
  attn_kernel<<<dim3(NTOK / 128, CB), 256, 0, stream>>>(tbf, pbf, gbf, ybf);
  out_kernel<<<dim3(NTOK / 16, CIN / 16, CB), 32, 0, stream>>>(
      ybf, w_w, w_b, x, (float*)d_out);
}
